// SNN_MODEL_20409684590692
// MI455X (gfx1250) — matrix operand order repaired and checked by reading, not compiled, not run on hardware
//
#include <hip/hip_runtime.h>

typedef __attribute__((ext_vector_type(16))) _Float16 v16h;
typedef __attribute__((ext_vector_type(8)))  _Float16 v8h;
typedef __attribute__((ext_vector_type(4)))  _Float16 v4h;
typedef __attribute__((ext_vector_type(8)))  float    v8f;

#define T_STEPS 100
#define B_TILE  16
#define THREADS 128          // 4 waves of 32
#define L1N     7488         // 16 * 13*6*6
#define NROWS   1248         // 16 * 13 * 6  (one row = 6 x-elements)
#define K2      208          // 13*4*4
#define NCHUNK  7            // ceil(208/32)

// ---- dynamic LDS layout (bytes) ----
constexpr size_t OFF_X    = 0;                    // 16*256 f32      = 16384
constexpr size_t OFF_W1   = OFF_X    + 16384;     // 208 f32         = 832
constexpr size_t OFF_B1   = OFF_W1   + 832;       // 13 f32 (pad)    = 64
constexpr size_t OFF_WF   = OFF_B1   + 64;        // 1152 f32        = 4608
constexpr size_t OFF_BF   = OFF_WF   + 4608;      // 2 f32 (pad)     = 16
constexpr size_t OFF_CUR1 = OFF_BF   + 16;        // 7488 f32        = 29952
constexpr size_t OFF_VI   = OFF_CUR1 + 29952;     // 7488 float2     = 59904
constexpr size_t OFF_S1W  = OFF_VI   + 59904;     // 16*13*6*12 f16  = 29952
constexpr size_t OFF_S2   = OFF_S1W  + 29952;     // 16*576 f16      = 18432
constexpr size_t SMEM_BYTES = OFF_S2 + 18432;     // 160144 B

__global__ void __launch_bounds__(THREADS)
snn_persistent_kernel(const float* __restrict__ gx,
                      const float* __restrict__ gw1,
                      const float* __restrict__ gb1,
                      const float* __restrict__ gw2,
                      const float* __restrict__ gb2,
                      const float* __restrict__ gwf,
                      const float* __restrict__ gbf,
                      float* __restrict__ out, int Btot)
{
    extern __shared__ char smem[];
    float*    lds_x    = (float*)   (smem + OFF_X);
    float*    lds_w1   = (float*)   (smem + OFF_W1);
    float*    lds_b1   = (float*)   (smem + OFF_B1);
    float*    lds_wf   = (float*)   (smem + OFF_WF);
    float*    lds_bf   = (float*)   (smem + OFF_BF);
    float*    lds_cur1 = (float*)   (smem + OFF_CUR1);
    float2*   lds_vi   = (float2*)  (smem + OFF_VI);    // (v1, i1) packed
    _Float16* lds_s1w  = (_Float16*)(smem + OFF_S1W);   // windowed spikes [m][c][y][12]
    _Float16* lds_spk2 = (_Float16*)(smem + OFF_S2);

    const int tid  = threadIdx.x;
    const int lane = tid & 31;
    const int wid  = tid >> 5;                 // wave id 0..3 -> channel group
    const int bblk = blockIdx.x * B_TILE;      // first batch of tile

    // ---------------- phase 0: stage inputs ----------------
    for (int i = tid; i < B_TILE * 256; i += THREADS) lds_x[i] = gx[bblk * 256 + i];
    for (int i = tid; i < 208;  i += THREADS) lds_w1[i] = gw1[i];
    if (tid < 13) lds_b1[tid] = gb1[tid];
    for (int i = tid; i < 1152; i += THREADS) lds_wf[i] = gwf[i];
    if (tid < 2)  lds_bf[tid] = gbf[tid];
    for (int i = tid; i < L1N;  i += THREADS) lds_vi[i] = make_float2(0.f, 0.f);
    __syncthreads();

    // conv1 + maxpool2 -> cur1 (time-invariant, computed once, lives in LDS)
    for (int idx = tid; idx < L1N; idx += THREADS) {
        int m = idx / 468, rem = idx % 468;
        int c = rem / 36,  q   = rem % 36;
        int y = q / 6,     xo  = q % 6;
        const float* xb = lds_x  + m * 256;
        const float* wc = lds_w1 + c * 16;
        float mx = -3.0e38f;
        for (int dy = 0; dy < 2; ++dy)
            for (int dx = 0; dx < 2; ++dx) {
                int oy = 2 * y + dy, ox = 2 * xo + dx;   // conv coords (<=14)
                float s = 0.f;
                #pragma unroll
                for (int ky = 0; ky < 4; ++ky)
                    #pragma unroll
                    for (int kx = 0; kx < 4; ++kx)
                        s += xb[(oy + ky) * 16 + ox + kx] * wc[ky * 4 + kx];
                mx = fmaxf(mx, s);
            }
        lds_cur1[idx] = mx + lds_b1[c];
    }

    // ---------------- conv2 weights -> resident B fragments ----------------
    // B layout (32x16 f16, K x N): lane n = col; lanes0-15 K={0-7,16-23}, lanes16-31 K={8-15,24-31}
    const int n    = lane & 15;
    const int ch   = wid * 16 + n;             // this wave's output channel
    const int kb2  = (lane >> 4) * 8;
    v16h bfrag[NCHUNK];
    #pragma unroll
    for (int cI = 0; cI < NCHUNK; ++cI) {
        #pragma unroll
        for (int h = 0; h < 16; ++h) {
            int K = cI * 32 + kb2 + (h & 7) + (h >> 3) * 16;
            float v = (K < K2) ? gw2[ch * K2 + K] : 0.f;
            bfrag[cI][h] = (_Float16)v;
        }
    }
    const float bias_ch = gb2[ch];

    // LIF-2 state resident in WMMA C-fragment layout: 9 positions x v8f
    v8f v2s[9], i2s[9];
    #pragma unroll
    for (int p = 0; p < 9; ++p) { v2s[p] = v8f{}; i2s[p] = v8f{}; }

    // LIF-3 state: thread (tid&3)==0 owns (batch m=tid>>3, output o=(tid>>2)&1)
    float v3r = 0.f, i3r = 0.f, spkSum = 0.f;

    const int mA  = lane & 15;                 // A/C batch row within tile
    const int kbA = (lane >> 4) * 8;           // A-fragment K base (0 or 8)
    // ky = kbA>>2 absorbed: all A-loads are immediate offsets off this base
    const _Float16* bp = lds_s1w + mA * 936 + kbA * 3;

    __syncthreads();

    // ================= time loop (fully on-chip) =================
    for (int t = 0; t < T_STEPS; ++t) {
        // ---- layer 1 LIF, row-wise; spikes stored pre-windowed ----
        for (int row = tid; row < NROWS; row += THREADS) {
            const int m   = row / 78;
            const int rem = row % 78;          // c*6 + y
            const int fl  = row * 6;           // flat element base
            float2* vi = lds_vi + fl;
            const float* cur = lds_cur1 + fl;
            _Float16 zh[6];
            #pragma unroll
            for (int x = 0; x < 6; ++x) {
                float2 s = vi[x];
                float vd = 0.9f * s.x + 0.1f * s.y;
                bool  z  = vd > 1.0f;
                s.x = z ? 0.0f : vd;
                s.y = 0.8f * s.y + cur[x];
                vi[x] = s;
                zh[x] = (_Float16)(z ? 1.0f : 0.0f);
            }
            // 3 overlapping 4-wide windows, 3 aligned b64 stores
            _Float16* sp = lds_s1w + m * 936 + rem * 12;
            *(v4h*)(sp + 0) = v4h{zh[0], zh[1], zh[2], zh[3]};
            *(v4h*)(sp + 4) = v4h{zh[1], zh[2], zh[3], zh[4]};
            *(v4h*)(sp + 8) = v4h{zh[2], zh[3], zh[4], zh[5]};
        }
        __syncthreads();

        // ---- layer 2: conv as 9x WMMA GEMM [16 x 208] x [208 x 16] ----
        #pragma unroll
        for (int p = 0; p < 9; ++p) {
            const int py = p / 3, px = p % 3;
            v8f acc;
            #pragma unroll
            for (int r = 0; r < 8; ++r) acc[r] = bias_ch;
            #pragma unroll
            for (int cI = 0; cI < NCHUNK; ++cI) {
                // halves 0-7:  channel c0=2cI,  rows py+ky, py+ky+1, window px
                // halves 8-15: channel c1=2cI+1 (zero tile when c1 == 13)
                const int c0 = 2 * cI, c1 = 2 * cI + 1;
                const int o0 = (c0 * 6 + py) * 12 + px * 4;
                v4h a0 = *(const v4h*)(bp + o0);
                v4h a1 = *(const v4h*)(bp + o0 + 12);
                v4h a2 = v4h{}, a3 = v4h{};
                if (c1 < 13) {
                    const int o1 = (c1 * 6 + py) * 12 + px * 4;
                    a2 = *(const v4h*)(bp + o1);
                    a3 = *(const v4h*)(bp + o1 + 12);
                }
                v8h lo = __builtin_shufflevector(a0, a1, 0,1,2,3,4,5,6,7);
                v8h hi = __builtin_shufflevector(a2, a3, 0,1,2,3,4,5,6,7);
                v16h a = __builtin_shufflevector(lo, hi,
                           0,1,2,3,4,5,6,7,8,9,10,11,12,13,14,15);
                acc = __builtin_amdgcn_wmma_f32_16x16x32_f16(
                          false, a, false, bfrag[cI], (short)0, acc, false, false);
            }
            // ---- layer 2 LIF directly in accumulator layout ----
            #pragma unroll
            for (int r = 0; r < 8; ++r) {
                float v = v2s[p][r], i = i2s[p][r];
                float vd = 0.9f * v + 0.1f * i;
                bool  z  = vd > 1.0f;
                v2s[p][r] = z ? 0.0f : vd;
                i2s[p][r] = 0.8f * i + acc[r];
                int m = (lane >> 4) * 8 + r;   // C layout: row
                lds_spk2[m * 576 + ch * 9 + p] = (_Float16)(z ? 1.0f : 0.0f);
            }
        }
        __syncthreads();

        // ---- layer 3: FC 576 -> 2 + LIF, 4 lanes per (batch, output) ----
        {
            const int m  = tid >> 3;
            const int o  = (tid >> 2) & 1;
            const int qq = tid & 3;
            // 16B-aligned slices: m*576+qq*144 halves, o*576+qq*144 floats
            const v8h*    s2v = (const v8h*)   (lds_spk2 + m * 576 + qq * 144);
            const float4* wv  = (const float4*)(lds_wf   + o * 576 + qq * 144);
            float s = 0.f;
            #pragma unroll
            for (int j8 = 0; j8 < 18; ++j8) {      // 144 = 18 * 8
                v8h    sv = s2v[j8];
                float4 wa = wv[2 * j8];
                float4 wb = wv[2 * j8 + 1];
                s += (float)sv[0] * wa.x + (float)sv[1] * wa.y +
                     (float)sv[2] * wa.z + (float)sv[3] * wa.w +
                     (float)sv[4] * wb.x + (float)sv[5] * wb.y +
                     (float)sv[6] * wb.z + (float)sv[7] * wb.w;
            }
            s += __shfl_xor(s, 1);
            s += __shfl_xor(s, 2);
            if (qq == 0) {
                float cur3 = s + lds_bf[o];
                float vd = 0.9f * v3r + 0.1f * i3r;
                bool  z  = vd > 1.0f;
                v3r = z ? 0.0f : vd;
                i3r = 0.8f * i3r + cur3;
                spkSum += z ? 1.0f : 0.0f;
                // v3_seq record: [T, B, 2] after spk_out [B,2]
                out[Btot * 2 + (t * Btot + (bblk + m)) * 2 + o] = v3r;
            }
        }
        __syncthreads();
    }

    // ---- final spike counts [B,2] ----
    {
        const int m  = tid >> 3;
        const int o  = (tid >> 2) & 1;
        const int qq = tid & 3;
        if (qq == 0) out[(bblk + m) * 2 + o] = spkSum;
    }
}

extern "C" void kernel_launch(void* const* d_in, const int* in_sizes, int n_in,
                              void* d_out, int out_size, void* d_ws, size_t ws_size,
                              hipStream_t stream) {
    (void)n_in; (void)out_size; (void)d_ws; (void)ws_size;
    const float* x  = (const float*)d_in[0];
    const float* w1 = (const float*)d_in[1];
    const float* b1 = (const float*)d_in[2];
    const float* w2 = (const float*)d_in[3];
    const float* b2 = (const float*)d_in[4];
    const float* wf = (const float*)d_in[5];
    const float* bf = (const float*)d_in[6];
    float* out = (float*)d_out;

    const int Btot    = in_sizes[0] / 256;     // x is [B,1,16,16]
    const int nblocks = Btot / B_TILE;         // 128 blocks of 16 batches

    snn_persistent_kernel<<<nblocks, THREADS, SMEM_BYTES, stream>>>(
        x, w1, b1, w2, b2, wf, bf, out, Btot);
}
